// BasicModel_39247411150867
// MI455X (gfx1250) — matrix operand order repaired and checked by reading, not compiled, not run on hardware
//
#include <hip/hip_runtime.h>
#include <hip/hip_bf16.h>

typedef _Float16 f16;
typedef __attribute__((ext_vector_type(16))) _Float16 v16h;
typedef __attribute__((ext_vector_type(2)))  _Float16 h2v;
typedef __attribute__((ext_vector_type(8)))  float    v8f;

#define B_    32
#define N_    64
#define K_    32
#define E_    256
#define DA_   128
#define HEAD_ 4
#define NHID_ 256
#define H3_   768
#define DIN_  1024
#define MLPH_ 512

__device__ __forceinline__ v8f wmma16(v16h a, v16h b, v8f c) {
  // D = A(16x32 f16) * B(32x16 f16) + C(16x16 f32)
  return __builtin_amdgcn_wmma_f32_16x16x32_f16(false, a, false, b, (short)0, c,
                                                false, false);
}

// Async global->LDS 16B transfer (CDNA5, ASYNCcnt-tracked).
__device__ __forceinline__ void async_g2l_b128(unsigned lds_off, const void* gaddr) {
  asm volatile("global_load_async_to_lds_b128 %0, %1, off"
               :: "v"(lds_off), "v"(gaddr) : "memory");
}
__device__ __forceinline__ void wait_async0() {
  asm volatile("s_wait_asynccnt 0x0" ::: "memory");
}

// A fragment (16x32, f16) from row-major f16 buffer, row block row0, k block k0.
__device__ __forceinline__ v16h load_a_f16(const f16* X, int ld, int row0, int k0, int lane) {
  int m = lane & 15, g = lane >> 4;
  v16h a;
#pragma unroll
  for (int v = 0; v < 8; ++v) {
    int k = ((v >> 2) << 4) + (g << 3) + ((v & 3) << 1);
    h2v p = *(const h2v*)(X + (size_t)(row0 + m) * ld + k0 + k);
    a[2 * v] = p[0]; a[2 * v + 1] = p[1];
  }
  return a;
}

// A fragment from row-major fp32 buffer with inline f32->f16 convert.
__device__ __forceinline__ v16h load_a_f32(const float* X, int ld, int row0, int k0, int lane) {
  int m = lane & 15, g = lane >> 4;
  v16h a;
#pragma unroll
  for (int v = 0; v < 8; ++v) {
    int k = ((v >> 2) << 4) + (g << 3) + ((v & 3) << 1);
    const float* p = X + (size_t)(row0 + m) * ld + k0 + k;
    a[2 * v] = (f16)p[0]; a[2 * v + 1] = (f16)p[1];
  }
  return a;
}

// B fragment (32x16) where B[k][n] = W[n0+n][k0+k]; W row-major [O][ldw] f16.
__device__ __forceinline__ v16h load_bT_f16(const f16* W, int ldw, int n0, int k0, int lane) {
  int n = lane & 15;
  int g = lane >> 4;
  v16h b;
#pragma unroll
  for (int v = 0; v < 8; ++v) {
    int k = ((v >> 2) << 4) + (g << 3) + ((v & 3) << 1);
    h2v p = *(const h2v*)(W + (size_t)(n0 + n) * ldw + k0 + k);
    b[2 * v] = p[0]; b[2 * v + 1] = p[1];
  }
  return b;
}

// ---------------------------------------------------------------- weights cvt
__global__ __launch_bounds__(256) void k_f32_to_f16(const float* __restrict__ s,
                                                    f16* __restrict__ d, int n) {
  int i = blockIdx.x * 256 + threadIdx.x;
  if (i < n) d[i] = (f16)s[i];
}

// ------------------------------------------------- stage 1: news attention
__global__ __launch_bounds__(256) void k_news_attn(
    const float* __restrict__ x, const int* __restrict__ len_li,
    const f16* __restrict__ Wna1h, const float* __restrict__ Wna2,
    float* __restrict__ newsF, f16* __restrict__ newsH) {
  __shared__ float Xs[K_][E_];      // 32 KB (async DMA target)
  __shared__ f16   Xh[K_][E_];      // 16 KB f16 copy for WMMA A
  __shared__ f16   A1[K_][DA_];     // 8 KB
  __shared__ float Sc[K_][HEAD_];
  __shared__ float Wt[K_][HEAD_];

  int bn = blockIdx.x;              // b*N + n
  int tid = threadIdx.x, lane = tid & 31, wave = tid >> 5;
  const float* xp = x + (size_t)bn * K_ * E_;

  // Async DMA the 32KB x-tile straight into LDS (8 x b128 per thread).
  unsigned ldsBase = (unsigned)(uintptr_t)(&Xs[0][0]);
  const char* gsrc = (const char*)xp;
#pragma unroll
  for (int j = 0; j < 8; ++j) {
    int i = j * 256 + tid;          // 2048 x 16B = 32KB
    async_g2l_b128(ldsBase + i * 16, gsrc + (size_t)i * 16);
  }
  wait_async0();
  __syncthreads();
  // One-time f32 -> f16 tile conversion (keeps cvt out of the WMMA k-loop).
  for (int i = tid; i < K_ * E_; i += 256)
    Xh[i >> 8][i & 255] = (f16)Xs[i >> 8][i & 255];
  __syncthreads();

  // a1 = tanh(X @ Wna1^T) : 2 x 8 tiles of 16x16, K=256
  for (int ti = wave * 2; ti < wave * 2 + 2; ++ti) {
    int mt = ti >> 3, nt = ti & 7;
    v8f c = {};
    for (int kk = 0; kk < E_; kk += 32) {
      v16h a = load_a_f16(&Xh[0][0], E_, mt * 16, kk, lane);
      v16h b = load_bT_f16(Wna1h, E_, nt * 16, kk, lane);
      c = wmma16(a, b, c);
    }
    int ml = (lane >> 4) << 3, nl = lane & 15;
#pragma unroll
    for (int r = 0; r < 8; ++r)
      A1[mt * 16 + r + ml][nt * 16 + nl] = (f16)tanhf(c[r]);
  }
  __syncthreads();

  // scores[k][h] = a1[k] . Wna2[h]
  if (tid < K_ * HEAD_) {
    int k = tid >> 2, h = tid & 3;
    float s = 0.f;
    for (int a = 0; a < DA_; ++a) s += (float)A1[k][a] * Wna2[h * DA_ + a];
    Sc[k][h] = s;
  }
  __syncthreads();

  // masked softmax over k
  if (tid < HEAD_) {
    int len = len_li[bn];
    float mx = -1e30f;
    for (int k = 0; k < len; ++k) mx = fmaxf(mx, Sc[k][tid]);
    float sum = 0.f;
    for (int k = 0; k < K_; ++k) {
      float e = (k < len) ? __expf(Sc[k][tid] - mx) : 0.f;
      Wt[k][tid] = e; sum += e;
    }
    float inv = 1.f / sum;
    for (int k = 0; k < K_; ++k) Wt[k][tid] *= inv;
  }
  __syncthreads();

  // news_M[h][e] = sum_k Wt[k][h] * X[k][e]
  int e = tid;
  float acc[HEAD_] = {0.f, 0.f, 0.f, 0.f};
  for (int k = 0; k < K_; ++k) {
    float xv = Xs[k][e];
#pragma unroll
    for (int h = 0; h < HEAD_; ++h) acc[h] += Wt[k][h] * xv;
  }
  size_t base = (size_t)bn * DIN_;
#pragma unroll
  for (int h = 0; h < HEAD_; ++h) {
    newsF[base + h * E_ + e] = acc[h];
    newsH[base + h * E_ + e] = (f16)acc[h];
  }
}

// ------------------------------------- stage 2a: bulk GRU input projection
__global__ __launch_bounds__(256) void k_gi_gemm(
    const f16* __restrict__ newsH, const f16* __restrict__ WihFh,
    const f16* __restrict__ WihBh, const float* __restrict__ bihF,
    const float* __restrict__ bihB, float* __restrict__ gi) {
  int dir = blockIdx.z;
  const f16* W = dir ? WihBh : WihFh;
  const float* bih = dir ? bihB : bihF;
  float* giD = gi + (size_t)dir * (B_ * N_) * H3_;
  int tid = threadIdx.x, lane = tid & 31, wave = tid >> 5;
  int m0 = blockIdx.x * 64 + (wave >> 1) * 16;
  int n0 = blockIdx.y * 96 + (wave & 1) * 48;
  v8f c0 = {}, c1 = {}, c2 = {};
  for (int kk = 0; kk < DIN_; kk += 32) {
    v16h a  = load_a_f16(newsH, DIN_, m0, kk, lane);
    v16h b0 = load_bT_f16(W, DIN_, n0,      kk, lane);
    v16h b1 = load_bT_f16(W, DIN_, n0 + 16, kk, lane);
    v16h b2 = load_bT_f16(W, DIN_, n0 + 32, kk, lane);
    c0 = wmma16(a, b0, c0); c1 = wmma16(a, b1, c1); c2 = wmma16(a, b2, c2);
  }
  int ml = (lane >> 4) << 3, nl = lane & 15;
#pragma unroll
  for (int r = 0; r < 8; ++r) {
    size_t row = (size_t)(m0 + r + ml) * H3_;
    giD[row + n0 + nl]      = c0[r] + bih[n0 + nl];
    giD[row + n0 + 16 + nl] = c1[r] + bih[n0 + 16 + nl];
    giD[row + n0 + 32 + nl] = c2[r] + bih[n0 + 32 + nl];
  }
}

// --------------------------------------------- stage 2b: recurrent GRU scan
__global__ __launch_bounds__(256) void k_gru(
    const float* __restrict__ hidden, const f16* __restrict__ WhhFh,
    const f16* __restrict__ WhhBh, const float* __restrict__ bhhF,
    const float* __restrict__ bhhB, const float* __restrict__ gi,
    float* __restrict__ out, float* __restrict__ hn) {
  __shared__ float Hs[B_][NHID_];   // 32 KB (fp32 master)
  __shared__ f16   Hh[B_][NHID_];   // 16 KB f16 mirror for WMMA A
  int dir = blockIdx.x;
  const f16* Whh = dir ? WhhBh : WhhFh;
  const float* bhh = dir ? bhhB : bhhF;
  const float* giD = gi + (size_t)dir * (B_ * N_) * H3_;
  int tid = threadIdx.x, lane = tid & 31, wave = tid >> 5;
  for (int i = tid; i < B_ * NHID_; i += 256) {
    float h0 = hidden[dir * B_ * NHID_ + i];
    Hs[i >> 8][i & 255] = h0;
    Hh[i >> 8][i & 255] = (f16)h0;
  }
  __syncthreads();

  int mt = wave >> 2;            // batch tile (0..1)
  int ntb = (wave & 3) * 4;      // 4 hidden-col tiles per wave
  int ml = (lane >> 4) << 3, nl = lane & 15;

  for (int t = 0; t < N_; ++t) {
    int day = dir ? (N_ - 1 - t) : t;
    float hnew[4][8];
#pragma unroll
    for (int j = 0; j < 4; ++j) {
      int n0 = (ntb + j) * 16;
      v8f cr = {}, cz = {}, cn = {};
      for (int kk = 0; kk < NHID_; kk += 32) {
        v16h a  = load_a_f16(&Hh[0][0], NHID_, mt * 16, kk, lane);
        v16h br = load_bT_f16(Whh, NHID_, n0,       kk, lane);
        v16h bz = load_bT_f16(Whh, NHID_, 256 + n0, kk, lane);
        v16h bn = load_bT_f16(Whh, NHID_, 512 + n0, kk, lane);
        cr = wmma16(a, br, cr); cz = wmma16(a, bz, cz); cn = wmma16(a, bn, cn);
      }
#pragma unroll
      for (int r = 0; r < 8; ++r) {
        int bb = mt * 16 + r + ml;
        int n  = n0 + nl;
        const float* gir = giD + ((size_t)bb * N_ + day) * H3_;
        float rg = 1.f / (1.f + __expf(-(gir[n]       + cr[r] + bhh[n])));
        float zg = 1.f / (1.f + __expf(-(gir[256 + n] + cz[r] + bhh[256 + n])));
        float ng = tanhf(gir[512 + n] + rg * (cn[r] + bhh[512 + n]));
        hnew[j][r] = (1.f - zg) * ng + zg * Hs[bb][n];
      }
    }
    __syncthreads();   // everyone done reading h for this step
#pragma unroll
    for (int j = 0; j < 4; ++j) {
#pragma unroll
      for (int r = 0; r < 8; ++r) {
        int bb = mt * 16 + r + ml;
        int n  = (ntb + j) * 16 + nl;
        Hs[bb][n] = hnew[j][r];
        Hh[bb][n] = (f16)hnew[j][r];
        out[((size_t)bb * N_ + day) * (2 * NHID_) + dir * NHID_ + n] = hnew[j][r];
      }
    }
    __syncthreads();
  }
  for (int i = tid; i < B_ * NHID_; i += 256)
    hn[dir * B_ * NHID_ + i] = Hs[i >> 8][i & 255];
}

// ------------------------------------------------ stage 3: day attention
__global__ __launch_bounds__(256) void k_day_attn(
    const float* __restrict__ out, const f16* __restrict__ Wda1h,
    const float* __restrict__ Wda2, float* __restrict__ daily,
    f16* __restrict__ dailyH) {
  __shared__ f16   D1[N_][DA_];     // 16 KB
  __shared__ float Sc[N_][HEAD_];
  __shared__ float Wt[N_][HEAD_];
  int b = blockIdx.x;
  const float* ob = out + (size_t)b * N_ * (2 * NHID_);
  int tid = threadIdx.x, lane = tid & 31, wave = tid >> 5;

  // d1 = tanh(out_b @ Wda1^T): 4 x 8 tiles, K=512
  for (int ti = wave * 4; ti < wave * 4 + 4; ++ti) {
    int mt = ti >> 3, nt = ti & 7;
    v8f c = {};
    for (int kk = 0; kk < 2 * NHID_; kk += 32) {
      v16h a = load_a_f32(ob, 2 * NHID_, mt * 16, kk, lane);
      v16h bb = load_bT_f16(Wda1h, 2 * NHID_, nt * 16, kk, lane);
      c = wmma16(a, bb, c);
    }
    int ml = (lane >> 4) << 3, nl = lane & 15;
#pragma unroll
    for (int r = 0; r < 8; ++r)
      D1[mt * 16 + r + ml][nt * 16 + nl] = (f16)tanhf(c[r]);
  }
  __syncthreads();

  { // dsc[n][h]
    int n = tid >> 2, h = tid & 3;
    float s = 0.f;
    for (int a = 0; a < DA_; ++a) s += (float)D1[n][a] * Wda2[h * DA_ + a];
    Sc[n][h] = s;
  }
  __syncthreads();
  if (tid < HEAD_) {
    float mx = -1e30f;
    for (int n = 0; n < N_; ++n) mx = fmaxf(mx, Sc[n][tid]);
    float sum = 0.f;
    for (int n = 0; n < N_; ++n) { float e = __expf(Sc[n][tid] - mx); Wt[n][tid] = e; sum += e; }
    float inv = 1.f / sum;
    for (int n = 0; n < N_; ++n) Wt[n][tid] *= inv;
  }
  __syncthreads();

  for (int j = 0; j < 8; ++j) {
    int idx = j * 256 + tid;             // [h*512 + d]
    int h = idx >> 9, d = idx & 511;
    float acc = 0.f;
    for (int n = 0; n < N_; ++n) acc += Wt[n][h] * ob[(size_t)n * (2 * NHID_) + d];
    daily[(size_t)b * 2048 + idx] = acc;
    dailyH[(size_t)b * 2048 + idx] = (f16)acc;
  }
}

// --------------------------------------------------- stage 4: MLP + decoder
__global__ __launch_bounds__(256) void k_head(
    const f16* __restrict__ dailyH, const f16* __restrict__ Wmlph,
    const float* __restrict__ bmlp, const float* __restrict__ Wdec,
    const float* __restrict__ bdec, float* __restrict__ dec) {
  __shared__ f16 M[B_][MLPH_];      // 32 KB
  int tid = threadIdx.x, lane = tid & 31, wave = tid >> 5;
  // mlp = relu(daily @ Wmlp^T + b): 2 x 32 tiles, K=2048
  for (int ti = wave * 8; ti < wave * 8 + 8; ++ti) {
    int mt = ti >> 5, nt = ti & 31;
    v8f c = {};
    for (int kk = 0; kk < 2048; kk += 32) {
      v16h a = load_a_f16(dailyH, 2048, mt * 16, kk, lane);
      v16h b = load_bT_f16(Wmlph, 2048, nt * 16, kk, lane);
      c = wmma16(a, b, c);
    }
    int ml = (lane >> 4) << 3, nl = lane & 15;
#pragma unroll
    for (int r = 0; r < 8; ++r) {
      int n = nt * 16 + nl;
      M[mt * 16 + r + ml][n] = (f16)fmaxf(c[r] + bmlp[n], 0.f);
    }
  }
  __syncthreads();
  if (tid < B_ * 2) {
    int b = tid >> 1, cls = tid & 1;
    float s = bdec[cls];
    for (int j = 0; j < MLPH_; ++j) s += (float)M[b][j] * Wdec[cls * MLPH_ + j];
    dec[b * 2 + cls] = s;
  }
}

extern "C" void kernel_launch(void* const* d_in, const int* in_sizes, int n_in,
                              void* d_out, int out_size, void* d_ws, size_t ws_size,
                              hipStream_t stream) {
  (void)in_sizes; (void)n_in; (void)out_size; (void)ws_size;
  const float* x      = (const float*)d_in[0];
  const int*   len_li = (const int*)d_in[1];
  const float* hidden = (const float*)d_in[2];
  const float* Wna1   = (const float*)d_in[3];
  const float* Wna2   = (const float*)d_in[4];
  const float* WihF   = (const float*)d_in[5];
  const float* WhhF   = (const float*)d_in[6];
  const float* bihF   = (const float*)d_in[7];
  const float* bhhF   = (const float*)d_in[8];
  const float* WihB   = (const float*)d_in[9];
  const float* WhhB   = (const float*)d_in[10];
  const float* bihB   = (const float*)d_in[11];
  const float* bhhB   = (const float*)d_in[12];
  const float* Wda1   = (const float*)d_in[13];
  const float* Wda2   = (const float*)d_in[14];
  const float* Wmlp   = (const float*)d_in[15];
  const float* bmlp   = (const float*)d_in[16];
  const float* Wdec   = (const float*)d_in[17];
  const float* bdec   = (const float*)d_in[18];

  char* ws = (char*)d_ws;
  size_t off = 0;
  auto take = [&](size_t bytes) -> void* {
    void* p = ws + off;
    off = (off + bytes + 255) & ~(size_t)255;
    return p;
  };

  f16*   Wna1h  = (f16*)take((size_t)DA_ * E_ * 2);
  f16*   WihFh  = (f16*)take((size_t)H3_ * DIN_ * 2);
  f16*   WihBh  = (f16*)take((size_t)H3_ * DIN_ * 2);
  f16*   WhhFh  = (f16*)take((size_t)H3_ * NHID_ * 2);
  f16*   WhhBh  = (f16*)take((size_t)H3_ * NHID_ * 2);
  f16*   Wda1h  = (f16*)take((size_t)DA_ * 2 * NHID_ * 2);
  f16*   Wmlph  = (f16*)take((size_t)MLPH_ * 2048 * 2);
  float* newsF  = (float*)take((size_t)B_ * N_ * DIN_ * 4);
  f16*   newsH  = (f16*)take((size_t)B_ * N_ * DIN_ * 2);
  float* gi     = (float*)take((size_t)2 * B_ * N_ * H3_ * 4);
  float* outBuf = (float*)take((size_t)B_ * N_ * 2 * NHID_ * 4);
  float* daily  = (float*)take((size_t)B_ * 2048 * 4);
  f16*   dailyH = (f16*)take((size_t)B_ * 2048 * 2);

  float* dec = (float*)d_out;        // [32,2]
  float* hn  = (float*)d_out + 64;   // [2,32,256]

#define CVT(src, dst, n) \
  k_f32_to_f16<<<((n) + 255) / 256, 256, 0, stream>>>(src, dst, n)
  CVT(Wna1, Wna1h, DA_ * E_);
  CVT(WihF, WihFh, H3_ * DIN_);
  CVT(WihB, WihBh, H3_ * DIN_);
  CVT(WhhF, WhhFh, H3_ * NHID_);
  CVT(WhhB, WhhBh, H3_ * NHID_);
  CVT(Wda1, Wda1h, DA_ * 2 * NHID_);
  CVT(Wmlp, Wmlph, MLPH_ * 2048);
#undef CVT

  k_news_attn<<<B_ * N_, 256, 0, stream>>>(x, len_li, Wna1h, Wna2, newsF, newsH);
  k_gi_gemm<<<dim3(32, 8, 2), 256, 0, stream>>>(newsH, WihFh, WihBh, bihF, bihB, gi);
  k_gru<<<2, 256, 0, stream>>>(hidden, WhhFh, WhhBh, bhhF, bhhB, gi, outBuf, hn);
  k_day_attn<<<B_, 256, 0, stream>>>(outBuf, Wda1h, Wda2, daily, dailyH);
  k_head<<<1, 256, 0, stream>>>(dailyH, Wmlph, bmlp, Wdec, bdec, dec);
}
